// BigGANSelfAttention_69002944577960
// MI455X (gfx1250) — matrix operand order repaired and checked by reading, not compiled, not run on hardware
//
#include <hip/hip_runtime.h>
#include <stdint.h>

// ---------------------------------------------------------------------------
// BigGAN self-attention for MI455X (gfx1250, wave32, WMMA bf16 16x16x32).
// B=16, C=512, H=W=64 (N=4096), CQK=64, CV=256, pool 2x2 -> M=1024.
// ---------------------------------------------------------------------------

typedef __bf16 bf16;
typedef __attribute__((ext_vector_type(16))) __bf16    v16bf;
typedef __attribute__((ext_vector_type(8)))  float     v8f;
typedef __attribute__((ext_vector_type(4)))  float     f32x4;
typedef __attribute__((ext_vector_type(4)))  uint32_t  u32x4;
typedef __attribute__((ext_vector_type(8)))  uint32_t  u32x8;

__device__ __forceinline__ uint16_t bf_bits(float f) {
  uint32_t u = __builtin_bit_cast(uint32_t, f);
  uint32_t r = u + 0x7FFFu + ((u >> 16) & 1u);   // round-to-nearest-even
  return (uint16_t)(r >> 16);
}
__device__ __forceinline__ bf16 f2bf(float f) {
  uint16_t b = bf_bits(f);
  return __builtin_bit_cast(bf16, b);
}
__device__ __forceinline__ uint32_t pack2(float lo, float hi) {
  return (uint32_t)bf_bits(lo) | ((uint32_t)bf_bits(hi) << 16);
}

// A fragment (16x32 bf16, M x K): base = row0 of [16][pitch] (K contiguous).
// Lane layout (ISA 7.12.2): lanes 0-15 row=lane, K 0-7 & 16-23; lanes 16-31 K 8-15 & 24-31.
__device__ __forceinline__ v16bf frag_a_bf16(const bf16* base, int pitch, int k0, int lane) {
  const int r = lane & 15, h = lane >> 4;
  const bf16* p = base + (size_t)r * pitch + k0 + h * 8;
  u32x4 lo = *(const u32x4*)p;
  u32x4 hi = *(const u32x4*)(p + 16);
  u32x8 w;
  w[0] = lo[0]; w[1] = lo[1]; w[2] = lo[2]; w[3] = lo[3];
  w[4] = hi[0]; w[5] = hi[1]; w[6] = hi[2]; w[7] = hi[3];
  return __builtin_bit_cast(v16bf, w);
}

// B fragment (32x16 bf16, K x N): colptr = &data[col][0] with K contiguous
// (caller already applied col = lane&15). lanes 0-15 hold K 0-7 & 16-23, 16-31 hold K 8-15 & 24-31.
__device__ __forceinline__ v16bf frag_b_bf16(const bf16* colptr, int k0, int lane) {
  const int h = lane >> 4;
  const bf16* p = colptr + k0 + h * 8;
  u32x4 lo = *(const u32x4*)p;
  u32x4 hi = *(const u32x4*)(p + 16);
  u32x8 w;
  w[0] = lo[0]; w[1] = lo[1]; w[2] = lo[2]; w[3] = lo[3];
  w[4] = hi[0]; w[5] = hi[1]; w[6] = hi[2]; w[7] = hi[3];
  return __builtin_bit_cast(v16bf, w);
}

__device__ __forceinline__ v8f wmma_bf16(v16bf a, v16bf b, v8f c) {
  return __builtin_amdgcn_wmma_f32_16x16x32_bf16(false, a, false, b, (short)0, c,
                                                 false, false);
}

// ---------------------------------------------------------------------------
// Prep: fp32 -> bf16 weight conversion (weights then live in L2).
// ---------------------------------------------------------------------------
__global__ void cvt_bf16_kernel(const float* __restrict__ src,
                                bf16* __restrict__ dst, int n) {
  int i = blockIdx.x * 256 + threadIdx.x;
  if (i < n) dst[i] = f2bf(src[i]);
}

// ---------------------------------------------------------------------------
// Stage 1: fused q/k/v projection + 2x2 maxpool of k,v.
// Grid: B*32 WGs of 256 threads. WG = (batch, 128-column n-tile) = 2 image rows
// -> contains whole pool windows. x panel staged once in LDS (130 KB bf16).
// ---------------------------------------------------------------------------
#define XP 520   // LDS pitch (halves) for xT, multiple of 8 for b128 alignment

__global__ __launch_bounds__(256) void qkv_pool_kernel(
    const float* __restrict__ hidden,                  // [B][512][4096]
    const bf16* __restrict__ Wq16, const bf16* __restrict__ Wk16,
    const bf16* __restrict__ Wv16,
    bf16* __restrict__ q16,                            // [B][4096][64]
    bf16* __restrict__ kp16,                           // [B][1024][64]
    bf16* __restrict__ vp16)                           // [B][256][1024]
{
  __shared__ __align__(16) bf16 xT[128 * XP];          // [n_local][c]
  __shared__ float pool_s[16 * 129];                   // [o][n_local]

  const int tid  = threadIdx.x;
  const int wave = tid >> 5, lane = tid & 31;
  const int l16  = lane & 15, hl = lane >> 4;
  const int b     = blockIdx.x >> 5;                   // 32 n-tiles per batch
  const int ntile = blockIdx.x & 31;
  const int n0    = ntile * 128;

  // Stage x[C=512][n0:n0+128] transposed into LDS as bf16 [n][c].
  const float* xsrc = hidden + (size_t)b * 512 * 4096 + n0;
  for (int idx = tid; idx < 512 * 128; idx += 256) {
    int c = idx >> 7, j = idx & 127;
    xT[j * XP + c] = f2bf(xsrc[(size_t)c * 4096 + j]);
  }
  __syncthreads();

  const int ncol = wave * 16 + l16;                    // this lane's n column
  const bf16* colp = &xT[(size_t)ncol * XP];

  for (int ot = 0; ot < 24; ++ot) {
    const bf16* W; int orow0, kind;                    // 0=q,1=k,2=v
    if (ot < 4)      { W = Wq16; orow0 = ot * 16;        kind = 0; }
    else if (ot < 8) { W = Wk16; orow0 = (ot - 4) * 16;  kind = 1; }
    else             { W = Wv16; orow0 = (ot - 8) * 16;  kind = 2; }

    v8f acc = {0.f, 0.f, 0.f, 0.f, 0.f, 0.f, 0.f, 0.f};
    const bf16* arow = W + (size_t)orow0 * 512;
    for (int kc = 0; kc < 16; ++kc) {                  // K = 512 in chunks of 32
      v16bf af = frag_a_bf16(arow, 512, kc * 32, lane);
      v16bf bfrag = frag_b_bf16(colp, kc * 32, lane);
      acc = wmma_bf16(af, bfrag, acc);
    }

    if (kind == 0) {
      // q, unpooled: lane holds col n = n0+ncol, rows o = orow0 + hl*8 + r
      u32x4 v;
      v[0] = pack2(acc[0], acc[1]); v[1] = pack2(acc[2], acc[3]);
      v[2] = pack2(acc[4], acc[5]); v[3] = pack2(acc[6], acc[7]);
      bf16* dst = q16 + ((size_t)b * 4096 + n0 + ncol) * 64 + orow0 + hl * 8;
      *(u32x4*)dst = v;
    } else {
      __syncthreads();                                 // pool_s reuse fence
      for (int r = 0; r < 8; ++r)
        pool_s[(hl * 8 + r) * 129 + ncol] = acc[r];
      __syncthreads();
      for (int t2 = tid; t2 < 512; t2 += 256) {        // 16 o x 32 pooled cols
        int o = t2 >> 5, p = t2 & 31;
        const float* row = &pool_s[o * 129];
        float m01 = fmaxf(row[2 * p],      row[2 * p + 1]);
        float m23 = fmaxf(row[64 + 2 * p], row[64 + 2 * p + 1]);
        bf16 mv = f2bf(fmaxf(m01, m23));
        int mg = ntile * 32 + p;                       // pooled index m
        if (kind == 1)
          kp16[((size_t)b * 1024 + mg) * 64 + orow0 + o] = mv;
        else
          vp16[((size_t)b * 256 + orow0 + o) * 1024 + mg] = mv;
      }
    }
  }
}

// ---------------------------------------------------------------------------
// Stage 2: per-16-row attention: scores -> softmax (LDS) -> P*V -> Wo + residual.
// Grid: B*256 WGs of 256 threads (8 waves). Score tile [16][1024] lives in LDS.
// ---------------------------------------------------------------------------
#define SP 1025   // f32 pitch for score tile
#define PP 1032   // bf16 pitch for exp(P) tile (multiple of 8)
#define AP 264    // bf16 pitch for attn tile  (multiple of 8)

__global__ __launch_bounds__(256) void attn_kernel(
    const float* __restrict__ hidden,
    const bf16* __restrict__ q16, const bf16* __restrict__ kp16,
    const bf16* __restrict__ vp16, const bf16* __restrict__ Wo16, // [512][256]
    const float* __restrict__ bo, const float* __restrict__ gating,
    float* __restrict__ out)
{
  __shared__ __align__(16) float s_lds[16 * SP];
  __shared__ __align__(16) bf16  p16[16 * PP];
  __shared__ __align__(16) bf16  a16[16 * AP];
  __shared__ float red[256];
  __shared__ float rowsum[16];

  const int tid  = threadIdx.x;
  const int wave = tid >> 5, lane = tid & 31;
  const int l16  = lane & 15, hl = lane >> 4;
  const int b  = blockIdx.x >> 8;
  const int n0 = (blockIdx.x & 255) * 16;
  const float g = gating[0];

  // ---- Phase 1: S[16][1024] = q(16x64) @ kp^T -------------------------------
  const bf16* qrow = q16 + ((size_t)b * 4096 + n0) * 64;
  v16bf qa0 = frag_a_bf16(qrow, 64, 0,  lane);
  v16bf qa1 = frag_a_bf16(qrow, 64, 32, lane);
  for (int it = 0; it < 8; ++it) {
    int m0 = (it * 8 + wave) * 16;
    const bf16* kcol = kp16 + ((size_t)b * 1024 + m0 + l16) * 64;
    v8f acc = {0.f, 0.f, 0.f, 0.f, 0.f, 0.f, 0.f, 0.f};
    acc = wmma_bf16(qa0, frag_b_bf16(kcol, 0,  lane), acc);
    acc = wmma_bf16(qa1, frag_b_bf16(kcol, 32, lane), acc);
    for (int r = 0; r < 8; ++r)
      s_lds[(hl * 8 + r) * SP + m0 + l16] = acc[r];
  }
  __syncthreads();

  // ---- Phase 2: softmax; store unnormalized exp as bf16, keep row sums -----
  {
    int row = tid >> 4, seg = tid & 15;                // 16 threads per row
    float* srow = &s_lds[row * SP + seg * 64];
    float lmax = -3.4e38f;
    for (int i = 0; i < 64; ++i) lmax = fmaxf(lmax, srow[i]);
    red[tid] = lmax;
    __syncthreads();
    float rmax = -3.4e38f;
    for (int j = 0; j < 16; ++j) rmax = fmaxf(rmax, red[row * 16 + j]);
    float lsum = 0.f;
    bf16* prow = &p16[row * PP + seg * 64];
    for (int i = 0; i < 64; ++i) {
      float e = __expf(srow[i] - rmax);
      lsum += e;
      prow[i] = f2bf(e);
    }
    __syncthreads();                                   // red reuse fence
    red[tid] = lsum;
    __syncthreads();
    if (seg == 0) {
      float s = 0.f;
      for (int j = 0; j < 16; ++j) s += red[row * 16 + j];
      rowsum[row] = s;
    }
  }
  __syncthreads();

  // ---- Phase 3: attn[16][256] = P @ vp^T, scaled by 1/rowsum ---------------
  for (int cvit = 0; cvit < 2; ++cvit) {
    int cv0 = cvit * 128 + wave * 16;
    const bf16* vcol = vp16 + ((size_t)b * 256 + cv0 + l16) * 1024;
    v8f acc = {0.f, 0.f, 0.f, 0.f, 0.f, 0.f, 0.f, 0.f};
    for (int kc = 0; kc < 32; ++kc) {                  // K = M = 1024
      __builtin_prefetch(vcol + (kc + 1) * 32, 0, 0);
      v16bf af = frag_a_bf16(p16, PP, kc * 32, lane);
      v16bf bfrag = frag_b_bf16(vcol, kc * 32, lane);
      acc = wmma_bf16(af, bfrag, acc);
    }
    for (int r = 0; r < 8; ++r) {
      int m = hl * 8 + r;
      a16[m * AP + cv0 + l16] = f2bf(acc[r] / rowsum[m]);
    }
  }
  __syncthreads();

  // ---- Phase 4: out = attn @ Wo^T + bo; residual + gating ------------------
  for (int cit = 0; cit < 4; ++cit) {
    int c0 = cit * 128 + wave * 16;
    int c = c0 + l16;
    const bf16* wcol = Wo16 + (size_t)c * 256;
    v8f acc = {0.f, 0.f, 0.f, 0.f, 0.f, 0.f, 0.f, 0.f};
    for (int kc = 0; kc < 8; ++kc) {                   // K = CV = 256
      v16bf af = frag_a_bf16(a16, AP, kc * 32, lane);
      v16bf bfrag = frag_b_bf16(wcol, kc * 32, lane);
      acc = wmma_bf16(af, bfrag, acc);
    }
    float bc = bo[c];
    size_t base = ((size_t)b * 512 + c) * 4096 + n0 + hl * 8;
    const f32x4* hv = (const f32x4*)(hidden + base);
    f32x4* ov = (f32x4*)(out + base);
    f32x4 h0 = hv[0], h1 = hv[1], o0, o1;
    for (int r = 0; r < 4; ++r) o0[r] = h0[r] + g * (acc[r] + bc);
    for (int r = 0; r < 4; ++r) o1[r] = h1[r] + g * (acc[4 + r] + bc);
    ov[0] = o0; ov[1] = o1;
  }
}

// ---------------------------------------------------------------------------
// Host launcher. Workspace layout (bytes): ~19.6 MB total.
// ---------------------------------------------------------------------------
extern "C" void kernel_launch(void* const* d_in, const int* in_sizes, int n_in,
                              void* d_out, int out_size, void* d_ws, size_t ws_size,
                              hipStream_t stream) {
  const float* hidden = (const float*)d_in[0];   // [16,512,64,64]
  const float* Wq     = (const float*)d_in[1];   // [64,512]
  const float* Wk     = (const float*)d_in[2];   // [64,512]
  const float* Wv     = (const float*)d_in[3];   // [256,512]
  const float* Wo     = (const float*)d_in[4];   // [512,256]
  const float* bo     = (const float*)d_in[5];   // [512]
  const float* gating = (const float*)d_in[6];   // scalar
  float* out = (float*)d_out;

  char* ws = (char*)d_ws;
  bf16* Wq16 = (bf16*)(ws + 0);                  //  64*512*2 =    65536
  bf16* Wk16 = (bf16*)(ws + 65536);              //  64*512*2 =    65536
  bf16* Wv16 = (bf16*)(ws + 131072);             // 256*512*2 =   262144
  bf16* Wo16 = (bf16*)(ws + 393216);             // 512*256*2 =   262144
  bf16* q16  = (bf16*)(ws + 655360);             // 16*4096*64*2 = 8388608
  bf16* kp16 = (bf16*)(ws + 9043968);            // 16*1024*64*2 = 2097152
  bf16* vp16 = (bf16*)(ws + 11141120);           // 16*256*1024*2 = 8388608

  cvt_bf16_kernel<<<(64 * 512 + 255) / 256, 256, 0, stream>>>(Wq, Wq16, 64 * 512);
  cvt_bf16_kernel<<<(64 * 512 + 255) / 256, 256, 0, stream>>>(Wk, Wk16, 64 * 512);
  cvt_bf16_kernel<<<(256 * 512 + 255) / 256, 256, 0, stream>>>(Wv, Wv16, 256 * 512);
  cvt_bf16_kernel<<<(512 * 256 + 255) / 256, 256, 0, stream>>>(Wo, Wo16, 512 * 256);

  qkv_pool_kernel<<<16 * 32, 256, 0, stream>>>(hidden, Wq16, Wk16, Wv16,
                                               q16, kp16, vp16);
  attn_kernel<<<16 * 256, 256, 0, stream>>>(hidden, q16, kp16, vp16, Wo16,
                                            bo, gating, out);
}
